// Cascade_CNN_RNN_33011118637662
// MI455X (gfx1250) — matrix operand order repaired and checked by reading, not compiled, not run on hardware
//
#include <hip/hip_runtime.h>

// =====================================================================
// Stochastic-computing CNN->MGU cascade for MI455X (gfx1250).
// 128 strictly-serial bit-cycles; per-cycle GEMMs on f16 WMMA
// (activations exactly +-1 -> exact in f16; fp32 accumulate).
// Weights converted once to f16 (~14MB, L2-resident across 128 reuses).
// All WMMA fragments fed by 128-bit vector loads; conv2 uses a staged
// im2col patch matrix so its A-operand is contiguous too.
// =====================================================================

typedef __attribute__((ext_vector_type(16))) _Float16 v16h;
typedef __attribute__((ext_vector_type(8)))  _Float16 v8h;
typedef __attribute__((ext_vector_type(8)))  float    v8f;

#define NCYC  32
#define WINN  4
#define BATCH 32
#define HH    10
#define WW    11
#define CHN   32
#define FC    1024
#define HID   1024
#define NCLS  5
#define RLIM  32.0f

#define S1 (BATCH*CHN*8*9)      /* 73728  conv1 out elems */
#define S2 (BATCH*2*CHN*6*7)    /* 86016  conv2 out elems */
#define S3 (BATCH*FC)           /* 32768 */
#define SH (BATCH*HID)          /* 32768 */
#define SO (BATCH*NCLS)         /* 160   */
#define M2 (BATCH*6*7)          /* 1344  conv2 im2col M   */
#define K2 (CHN*9)              /* 288   conv2 im2col K   */
#define K3 (2*CHN*6*7)          /* 2688  fc3 K            */

__device__ __forceinline__ int bitrev5(int c) {
  // van der Corput base-2 threshold (Sobol-1D), 5 bits
  return ((c & 1) << 4) | ((c & 2) << 2) | (c & 4) | ((c & 8) >> 2) | ((c & 16) >> 4);
}

// A-fragment: lane needs halves [0..7] and [16..23] from p (16B-aligned)
__device__ __forceinline__ v16h load_a16(const _Float16* __restrict__ p) {
  v8h lo = *(const v8h*)(p);
  v8h hi = *(const v8h*)(p + 16);
  return __builtin_shufflevector(lo, hi, 0,1,2,3,4,5,6,7,8,9,10,11,12,13,14,15);
}
// B-fragment: lane needs halves [0..7] and [16..23] from p (16B-aligned)
__device__ __forceinline__ v16h load_b16(const _Float16* __restrict__ p) {
  v8h lo = *(const v8h*)(p);
  v8h hi = *(const v8h*)(p + 16);
  return __builtin_shufflevector(lo, hi, 0,1,2,3,4,5,6,7,8,9,10,11,12,13,14,15);
}

// ---------------------------------------------------------------- convert
__global__ void k_f32_to_f16(const float* __restrict__ s, _Float16* __restrict__ d, int n) {
  int i = blockIdx.x * blockDim.x + threadIdx.x;
  if (i < n) d[i] = (_Float16)s[i];
}

__global__ void k_pad_fc5(const float* __restrict__ s, _Float16* __restrict__ d) {
  // fc5_w [5,1024] -> zero-padded [16,1024] so the WMMA B-loads need no guard
  int i = blockIdx.x * blockDim.x + threadIdx.x;
  if (i < 16 * HID) {
    int n = i / HID;
    d[i] = (n < NCLS) ? (_Float16)s[i] : (_Float16)0.f;
  }
}

// ---------------------------------------------------------------- window init
__global__ void k_win_init(int win,
                           float* a1, float* c1, float* a2, float* c2,
                           float* a3, float* c3, float* af, float* an,
                           float* hsum, int* hsrc, float* ao, float* osum) {
  int i = blockIdx.x * blockDim.x + threadIdx.x;
  if (i < S1) { a1[i] = 0.f; c1[i] = 0.f; }
  if (i < S2) { a2[i] = 0.f; c2[i] = 0.f; }
  if (i < S3) { a3[i] = 0.f; c3[i] = 0.f; }
  if (i < SH) {
    af[i] = 0.f; an[i] = 0.f;
    float hx = (win == 0) ? 0.f : (hsum[i] / (float)NCYC);   // hPEM estimate
    float q  = rintf((hx + 1.f) * 0.5f * (float)NCYC);       // SourceGen quant
    q = fminf(fmaxf(q, 0.f), (float)NCYC);
    hsrc[i] = (int)q;
    hsum[i] = 0.f;
  }
  if (i < SO) { ao[i] = 0.f; if (win == 0) osum[i] = 0.f; }
}

// ---------------------------------------------------------------- per-cycle hidden bits
__global__ void k_hbits(int cyc, const int* __restrict__ hsrc, _Float16* __restrict__ hbh) {
  int i = blockIdx.x * blockDim.x + threadIdx.x;
  if (i < SH) {
    int thr = bitrev5(cyc);
    hbh[i] = (_Float16)((hsrc[i] > thr) ? 1.f : -1.f);
  }
}

// ---------------------------------------------------------------- conv1 (tiny, VALU)
__global__ void k_conv1(int win, int cyc, const float* __restrict__ in,
                        const float* __restrict__ w1, const float* __restrict__ b1,
                        float* a1, float* c1, _Float16* r1) {
  int i = blockIdx.x * blockDim.x + threadIdx.x;
  if (i >= S1) return;
  int x = i % 9, y = (i / 9) % 8, o = (i / 72) % CHN, b = i / (72 * CHN);
  int   thr  = bitrev5(cyc);
  float zbit = (cyc & 1) ? -1.f : 1.f;
  const float* ip = in + ((b * WINN + win) * HH) * WW;
  const float* wp = w1 + o * 9;
  float s = 0.f;
#pragma unroll
  for (int r = 0; r < 3; ++r)
#pragma unroll
    for (int t = 0; t < 3; ++t) {
      float v  = ip[(y + r) * WW + (x + t)];
      float q  = fminf(fmaxf(rintf((v + 1.f) * 16.f), 0.f), 32.f); // BSGen source
      float xb = (q > (float)thr) ? 1.f : -1.f;                    // bit-serial input
      s += wp[r * 3 + t] * xb;
    }
  s += b1[o];
  float t0  = a1[i] + s;                       // FSU sigma-delta
  float bit = (t0 >= 0.f) ? 1.f : -1.f;
  a1[i] = t0 - bit;
  float c = fminf(fmaxf(c1[i] + bit, -RLIM), RLIM);  // FSUReLU counter
  c1[i] = c;
  r1[i] = (_Float16)((c >= 0.f) ? bit : zbit);
}

// ---------------------------------------------------------------- im2col repack (conv2 A)
// Apat[m][k] with m=(b,y,x), k=(c,i,j): makes conv2's A-operand contiguous.
__global__ void k_im2col(const _Float16* __restrict__ r1, _Float16* __restrict__ ap) {
  int i = blockIdx.x * blockDim.x + threadIdx.x;
  if (i >= M2 * K2) return;
  int m = i / K2, k = i % K2;
  int b = m / 42, p = m % 42, y = p / 7, x = p % 7;
  int c = k / 9, rr = (k % 9) / 3, ss = k % 3;
  ap[i] = r1[((b * CHN + c) * 8 + y + rr) * 9 + (x + ss)];
}

// ---------------------------------------------------------------- conv2 (im2col WMMA)
__global__ void __launch_bounds__(32)
k_conv2(int cyc, const _Float16* __restrict__ ap, const _Float16* __restrict__ w2,
        const float* __restrict__ b2, float* a2, float* c2, _Float16* r2) {
  int lane = threadIdx.x;
  int mt = blockIdx.x;                 // 0..83
  int nt = blockIdx.y;                 // 0..3
  float zbit = (cyc & 1) ? -1.f : 1.f;
  int mA = lane & 15, kbA = (lane & 16) ? 8 : 0;
  int n  = lane & 15, kbB = (lane & 16) ? 8 : 0;
  const _Float16* arow = ap + (mt * 16 + mA) * K2 + kbA;
  const _Float16* brow = w2 + (nt * 16 + n) * K2 + kbB;  // [64,288] row-major
  v8f acc = {};
#pragma unroll 3
  for (int kk = 0; kk < K2; kk += 32) {
    v16h A = load_a16(arow + kk);
    v16h B = load_b16(brow + kk);
    acc = __builtin_amdgcn_wmma_f32_16x16x32_f16(false, A, false, B, (short)0, acc, false, false);
  }
  int hi = (lane & 16) ? 8 : 0;
#pragma unroll
  for (int r = 0; r < 8; ++r) {
    int m = mt * 16 + r + hi;
    int o = nt * 16 + n;
    int bb = m / 42, pp = m % 42;
    int idx = (bb * (2 * CHN) + o) * 42 + pp;
    float y  = acc[r] + b2[o];
    float t0 = a2[idx] + y;
    float bit = (t0 >= 0.f) ? 1.f : -1.f;
    a2[idx] = t0 - bit;
    float c = fminf(fmaxf(c2[idx] + bit, -RLIM), RLIM);
    c2[idx] = c;
    r2[idx] = (_Float16)((c >= 0.f) ? bit : zbit);
  }
}

// ---------------------------------------------------------------- fc3 (WMMA)
__global__ void __launch_bounds__(32)
k_fc3(int cyc, const _Float16* __restrict__ r2, const _Float16* __restrict__ wt,
      const float* __restrict__ bias, float* a3, float* c3, _Float16* r3) {
  int lane = threadIdx.x;
  int nt = blockIdx.x;                 // 0..63
  int mt = blockIdx.y;                 // 0..1
  float zbit = (cyc & 1) ? -1.f : 1.f;
  int mA = lane & 15, kbA = (lane & 16) ? 8 : 0;
  int n  = lane & 15, kbB = (lane & 16) ? 8 : 0;
  const _Float16* arow = r2 + (mt * 16 + mA) * K3 + kbA;
  const _Float16* brow = wt + (size_t)(nt * 16 + n) * K3 + kbB;
  v8f acc = {};
#pragma unroll 2
  for (int kk = 0; kk < K3; kk += 32) {
    v16h A = load_a16(arow + kk);
    v16h B = load_b16(brow + kk);
    acc = __builtin_amdgcn_wmma_f32_16x16x32_f16(false, A, false, B, (short)0, acc, false, false);
  }
  int hi = (lane & 16) ? 8 : 0;
#pragma unroll
  for (int r = 0; r < 8; ++r) {
    int m = mt * 16 + r + hi, j = nt * 16 + n;
    int idx = m * FC + j;
    float y  = acc[r] + bias[j];
    float t0 = a3[idx] + y;
    float bit = (t0 >= 0.f) ? 1.f : -1.f;
    a3[idx] = t0 - bit;
    float c = fminf(fmaxf(c3[idx] + bit, -RLIM), RLIM);
    c3[idx] = c;
    r3[idx] = (_Float16)((c >= 0.f) ? bit : zbit);
  }
}

// ---------------------------------------------------------------- MGU (fused 4-tile WMMA)
// Each wave computes If/In (r3 x wih^T) and Hf/Hn (hb x whh^T) for one 16x16
// (batch, hidden) tile-pair, then does sigma-delta gating in-register.
__global__ void __launch_bounds__(32)
k_mgu(int cyc, const _Float16* __restrict__ r3, const _Float16* __restrict__ hbh,
      const _Float16* __restrict__ wih, const float* __restrict__ bih,
      const _Float16* __restrict__ whh, const float* __restrict__ bhh,
      float* af, float* an, float* hsum, _Float16* hob) {
  int lane = threadIdx.x;
  int nt = blockIdx.x;                 // 0..63 -> j0
  int mt = blockIdx.y;                 // 0..1
  int mA = lane & 15, kbA = (lane & 16) ? 8 : 0;
  int n  = lane & 15, kbB = (lane & 16) ? 8 : 0;
  int j0 = nt * 16;
  const _Float16* arow  = r3  + (mt * 16 + mA) * HID + kbA;
  const _Float16* ahrow = hbh + (mt * 16 + mA) * HID + kbA;
  const _Float16* bIf = wih + (size_t)(j0 + n) * HID + kbB;
  const _Float16* bIn = wih + (size_t)(HID + j0 + n) * HID + kbB;
  const _Float16* bHf = whh + (size_t)(j0 + n) * HID + kbB;
  const _Float16* bHn = whh + (size_t)(HID + j0 + n) * HID + kbB;
  v8f aIf = {}, aIn = {}, aHf = {}, aHn = {};
#pragma unroll 2
  for (int kk = 0; kk < HID; kk += 32) {
    v16h A  = load_a16(arow + kk);
    v16h Ah = load_a16(ahrow + kk);
    aIf = __builtin_amdgcn_wmma_f32_16x16x32_f16(false, A,  false, load_b16(bIf + kk), (short)0, aIf, false, false);
    aIn = __builtin_amdgcn_wmma_f32_16x16x32_f16(false, A,  false, load_b16(bIn + kk), (short)0, aIn, false, false);
    aHf = __builtin_amdgcn_wmma_f32_16x16x32_f16(false, Ah, false, load_b16(bHf + kk), (short)0, aHf, false, false);
    aHn = __builtin_amdgcn_wmma_f32_16x16x32_f16(false, Ah, false, load_b16(bHn + kk), (short)0, aHn, false, false);
  }
  int hi = (lane & 16) ? 8 : 0;
#pragma unroll
  for (int r = 0; r < 8; ++r) {
    int b = mt * 16 + r + hi, j = j0 + n;
    int idx = b * HID + j;
    float gf  = aIf[r] + aHf[r] + bih[j] + bhh[j];
    float t0  = af[idx] + gf;
    float fb  = (t0 >= 0.f) ? 1.f : -1.f;
    af[idx] = t0 - fb;
    bool fsel = fb > 0.f;                                    // stochastic gate
    float gn  = aIn[r] + bih[HID + j] + (fsel ? (aHn[r] + bhh[HID + j]) : 0.f);
    float u0  = an[idx] + gn;
    float nb  = (u0 >= 0.f) ? 1.f : -1.f;
    an[idx] = u0 - nb;
    float hbv = (float)hbh[idx];
    float ho  = fsel ? nb : hbv;                             // MUX f*n + (1-f)*h
    hsum[idx] += ho;
    hob[idx] = (_Float16)ho;
  }
}

// ---------------------------------------------------------------- fc5 (+oPEM), last window
__global__ void __launch_bounds__(64)
k_fc5(const _Float16* __restrict__ hob, const _Float16* __restrict__ w5,
      const float* __restrict__ b5, float* ao, float* osum) {
  int lane = threadIdx.x & 31;
  int mt   = threadIdx.x >> 5;         // 0..1
  int mA = lane & 15, kbA = (lane & 16) ? 8 : 0;
  int n  = lane & 15, kbB = (lane & 16) ? 8 : 0;
  const _Float16* arow = hob + (mt * 16 + mA) * HID + kbA;
  const _Float16* brow = w5 + n * HID + kbB;   // padded [16,1024]: no guard in loop
  v8f acc = {};
#pragma unroll 2
  for (int kk = 0; kk < HID; kk += 32) {
    v16h A = load_a16(arow + kk);
    v16h B = load_b16(brow + kk);
    acc = __builtin_amdgcn_wmma_f32_16x16x32_f16(false, A, false, B, (short)0, acc, false, false);
  }
  if (n < NCLS) {                        // divergence only after all WMMAs
    int hi = (lane & 16) ? 8 : 0;
#pragma unroll
    for (int r = 0; r < 8; ++r) {
      int m = mt * 16 + r + hi;
      int idx = m * NCLS + n;
      float y  = acc[r] + b5[n];
      float t0 = ao[idx] + y;
      float ob = (t0 >= 0.f) ? 1.f : -1.f;
      ao[idx] = t0 - ob;
      osum[idx] += ob;
    }
  }
}

// ---------------------------------------------------------------- finalize
__global__ void k_fin(const float* __restrict__ osum, float* __restrict__ out) {
  int i = blockIdx.x * blockDim.x + threadIdx.x;
  if (i < SO) out[i] = osum[i] / (float)NCYC;
}

// =====================================================================
extern "C" void kernel_launch(void* const* d_in, const int* in_sizes, int n_in,
                              void* d_out, int out_size, void* d_ws, size_t ws_size,
                              hipStream_t stream) {
  const float* input   = (const float*)d_in[0];
  const float* conv1_w = (const float*)d_in[1];
  const float* conv1_b = (const float*)d_in[2];
  const float* conv2_w = (const float*)d_in[3];
  const float* conv2_b = (const float*)d_in[4];
  const float* fc3_w   = (const float*)d_in[5];
  const float* fc3_b   = (const float*)d_in[6];
  const float* wih     = (const float*)d_in[7];
  const float* bih     = (const float*)d_in[8];
  const float* whh     = (const float*)d_in[9];
  const float* bhh     = (const float*)d_in[10];
  const float* fc5_w   = (const float*)d_in[11];
  const float* fc5_b   = (const float*)d_in[12];
  float* out = (float*)d_out;

  // bump allocator over d_ws (~17.3 MB total)
  char* p = (char*)d_ws;
  auto alloc = [&](size_t bytes) -> char* {
    char* q = p; p += (bytes + 255) & ~(size_t)255; return q;
  };
  _Float16* w2h  = (_Float16*)alloc((size_t)64 * K2 * 2);
  _Float16* fc3h = (_Float16*)alloc((size_t)FC * K3 * 2);
  _Float16* wihh = (_Float16*)alloc((size_t)2 * HID * FC * 2);
  _Float16* whhh = (_Float16*)alloc((size_t)2 * HID * HID * 2);
  _Float16* fc5h = (_Float16*)alloc((size_t)16 * HID * 2);
  _Float16* r1h  = (_Float16*)alloc((size_t)S1 * 2);
  _Float16* apat = (_Float16*)alloc((size_t)M2 * K2 * 2);
  _Float16* r2h  = (_Float16*)alloc((size_t)S2 * 2);
  _Float16* r3h  = (_Float16*)alloc((size_t)S3 * 2);
  _Float16* hbh  = (_Float16*)alloc((size_t)SH * 2);
  _Float16* hobh = (_Float16*)alloc((size_t)SH * 2);
  float* a1 = (float*)alloc((size_t)S1 * 4); float* c1 = (float*)alloc((size_t)S1 * 4);
  float* a2 = (float*)alloc((size_t)S2 * 4); float* c2 = (float*)alloc((size_t)S2 * 4);
  float* a3 = (float*)alloc((size_t)S3 * 4); float* c3 = (float*)alloc((size_t)S3 * 4);
  float* af = (float*)alloc((size_t)SH * 4); float* an = (float*)alloc((size_t)SH * 4);
  float* hs = (float*)alloc((size_t)SH * 4); int*   hq = (int*)alloc((size_t)SH * 4);
  float* ao = (float*)alloc((size_t)SO * 4); float* os = (float*)alloc((size_t)SO * 4);

  // one-time weight conversion (L2-resident across all 128 cycle reuses)
  auto cvt = [&](const float* s, _Float16* d, int n) {
    k_f32_to_f16<<<(n + 255) / 256, 256, 0, stream>>>(s, d, n);
  };
  cvt(conv2_w, w2h, 64 * K2);
  cvt(fc3_w,   fc3h, FC * K3);
  cvt(wih,     wihh, 2 * HID * FC);
  cvt(whh,     whhh, 2 * HID * HID);
  k_pad_fc5<<<(16 * HID + 255) / 256, 256, 0, stream>>>(fc5_w, fc5h);

  for (int win = 0; win < WINN; ++win) {
    k_win_init<<<(S2 + 255) / 256, 256, 0, stream>>>(win, a1, c1, a2, c2, a3, c3,
                                                     af, an, hs, hq, ao, os);
    for (int cyc = 0; cyc < NCYC; ++cyc) {
      k_conv1<<<(S1 + 255) / 256, 256, 0, stream>>>(win, cyc, input, conv1_w, conv1_b,
                                                    a1, c1, r1h);
      k_hbits<<<(SH + 255) / 256, 256, 0, stream>>>(cyc, hq, hbh);
      k_im2col<<<(M2 * K2 + 255) / 256, 256, 0, stream>>>(r1h, apat);
      { dim3 g(84, 4); k_conv2<<<g, 32, 0, stream>>>(cyc, apat, w2h, conv2_b, a2, c2, r2h); }
      { dim3 g(64, 2); k_fc3 <<<g, 32, 0, stream>>>(cyc, r2h, fc3h, fc3_b, a3, c3, r3h); }
      { dim3 g(64, 2); k_mgu <<<g, 32, 0, stream>>>(cyc, r3h, hbh, wihh, bih, whhh, bhh,
                                                    af, an, hs, hobh); }
      if (win == WINN - 1)
        k_fc5<<<1, 64, 0, stream>>>(hobh, fc5h, fc5_b, ao, os);
    }
  }
  k_fin<<<1, 256, 0, stream>>>(os, out);
}
